// TransformerModel_21517786153303
// MI455X (gfx1250) — matrix operand order repaired and checked by reading, not compiled, not run on hardware
//
#include <hip/hip_runtime.h>
#include <cstddef>
#include <cstdint>

// ---------------- problem constants ----------------
#define LAYERS 4
#define BBATCH 2
#define SS 2048
#define FF 256
#define HH 8
#define DKK 32
#define HD 256            // H*DK
#define FFN 1024          // 4*F
#define BS 4096           // B*S
#define REL 21            // 2M+1

typedef __attribute__((ext_vector_type(16))) __bf16 bf16x16;
typedef __attribute__((ext_vector_type(8)))  __bf16 bf16x8;
typedef __attribute__((ext_vector_type(2)))  __bf16 bf16x2;
typedef __attribute__((ext_vector_type(8)))  float  v8f;

__device__ __forceinline__ v8f wmma_bf16(bf16x16 a, bf16x16 b, v8f c) {
  return __builtin_amdgcn_wmma_f32_16x16x32_bf16(false, a, false, b, (short)0, c,
                                                 false, false);
}
__device__ __forceinline__ bf16x16 cat16(bf16x8 lo, bf16x8 hi) {
  return __builtin_shufflevector(lo, hi, 0, 1, 2, 3, 4, 5, 6, 7,
                                 8, 9, 10, 11, 12, 13, 14, 15);
}

// ---------------- f32 -> bf16 convert ----------------
__global__ void cvt_kernel(const float* __restrict__ in, __bf16* __restrict__ out, int n) {
  int i = blockIdx.x * blockDim.x + threadIdx.x;
  if (i < n) out[i] = (__bf16)in[i];
}

// ---------------- bf16-WMMA GEMM, 16x64 strip per wave ----------------
// A: [Mrows,K] bf16 row-major, W: [K,N] bf16 row-major.
// mode bits: 1=GELU, 2=write f32 Cf[M,N], 4=write bf16 Cb[M,N],
//            8=bf16 Cb as QKV-pack [B,H,S,DK], 16=bf16 Cb as K^T-pack [B,H,DK,S].
__global__ __launch_bounds__(256) void gemm_kernel(
    const __bf16* __restrict__ A, const __bf16* __restrict__ W,
    const float* __restrict__ bias, float* __restrict__ Cf,
    __bf16* __restrict__ Cb, int Mrows, int N, int K, int mode) {
  const int lane = threadIdx.x & 31;
  const int wave = threadIdx.x >> 5;
  const int half = lane >> 4;
  const int nn   = lane & 15;
  const int m0 = (blockIdx.y * 8 + wave) * 16;
  const int n0 = blockIdx.x * 64;                   // 4 N-tiles per wave
  const int kr = nn + 16 * half;                    // B fragment K-row
  const __bf16* arow = A + (size_t)(m0 + nn) * K;   // A fragment M-row

  v8f acc[4] = {};
  for (int k0 = 0; k0 < K; k0 += 32) {
    // unconditional speculative prefetch: straight-line loop body
    __builtin_prefetch(arow + k0 + 32, 0, 1);                        // global_prefetch_b8
    __builtin_prefetch(W + (size_t)(k0 + 32 + kr) * N + n0, 0, 1);
    bf16x8 a0 = *(const bf16x8*)(arow + k0 + 8 * half);              // global_load_b128
    bf16x8 a1 = *(const bf16x8*)(arow + k0 + 16 + 8 * half);
    bf16x16 af = cat16(a0, a1);
    const __bf16* wrow = W + (size_t)k0 * N + n0 + nn;
#pragma unroll
    for (int t = 0; t < 4; ++t) {                    // 4 independent WMMAs, A reused
      bf16x16 bfv;
#pragma unroll
      for (int s = 0; s < 16; ++s)
        bfv[s] = wrow[(size_t)(8 * half + s + (s & 8)) * N + 16 * t];
      acc[t] = wmma_bf16(af, bfv, acc[t]);
    }
  }
#pragma unroll
  for (int t = 0; t < 4; ++t) {
#pragma unroll
    for (int r = 0; r < 8; ++r) {
      int row = m0 + r + 8 * half;
      int col = n0 + 16 * t + nn;
      float cv = acc[t][r] + bias[col];
      if (mode & 1) cv = 0.5f * cv * (1.0f + erff(cv * 0.70710678118654752f));
      if (mode & 2) Cf[(size_t)row * N + col] = cv;
      if (mode & 4) Cb[(size_t)row * N + col] = (__bf16)cv;
      if (mode & 8) {   // [B,H,S,DK]
        int b = row >> 11, s = row & (SS - 1), h = col >> 5, d = col & (DKK - 1);
        Cb[(((size_t)b * HH + h) * SS + s) * DKK + d] = (__bf16)cv;
      }
      if (mode & 16) {  // [B,H,DK,S]
        int b = row >> 11, s = row & (SS - 1), h = col >> 5, d = col & (DKK - 1);
        Cb[(((size_t)b * HH + h) * DKK + d) * SS + s] = (__bf16)cv;
      }
    }
  }
}

// ---------------- fused relative-position attention ----------------
// qp,vp: [B,H,S,DK] bf16; kt: [B,H,DK,S] bf16. ctxb: [B*S, H*DK] bf16.
// asum accumulates attn/L (store on first layer, += after).
__global__ __launch_bounds__(128) void attn_kernel(
    const __bf16* __restrict__ qp, const __bf16* __restrict__ kt,
    const __bf16* __restrict__ vp, const int* __restrict__ pos,
    const int* __restrict__ maskp, const float* __restrict__ relk,
    const float* __restrict__ relb, __bf16* __restrict__ ctxb,
    float* __restrict__ asum, int first) {
  __shared__ __bf16 plds[4][16 * 32];
  const int lane = threadIdx.x & 31;
  const int wave = threadIdx.x >> 5;
  const int half = lane >> 4;
  const int nn   = lane & 15;
  const int b = blockIdx.y / HH, h = blockIdx.y % HH;
  const int i0 = (blockIdx.x * 4 + wave) * 16;
  const float scale = 0.17677669529663687f;  // 1/sqrt(32)
  const float NEGV  = -4294967295.0f;        // -2^32+1
  const size_t bhS = ((size_t)b * HH + h) * SS;

  // Q fragment (A-layout), DK==32 == WMMA K: two b128 loads total.
  const __bf16* qrow = qp + (bhS + i0 + nn) * DKK;
  bf16x16 qf = cat16(*(const bf16x8*)(qrow + 8 * half),
                     *(const bf16x8*)(qrow + 16 + 8 * half));

  float pi[8];
#pragma unroll
  for (int r = 0; r < 8; ++r) pi[r] = (float)pos[b * SS + i0 + r + 8 * half];

  const __bf16* krow = kt + (((size_t)b * HH + h) * DKK) * SS;

  auto score_tile = [&](int j0, float* out) {
    bf16x16 kf;
#pragma unroll
    for (int s = 0; s < 16; ++s)
      kf[s] = krow[(size_t)(8 * half + s + (s & 8)) * SS + j0 + nn];
    v8f c = {};
    c = wmma_bf16(qf, kf, c);
    float pj = (float)pos[b * SS + j0 + nn];
    int   ok = maskp[b * SS + j0 + nn];
#pragma unroll
    for (int r = 0; r < 8; ++r) {
      float dd = fabsf(pi[r] - pj);
      if (dd > 10.0f) dd = 10.0f + log2f(dd - 10.0f);
      dd = fminf(dd, 20.0f);
      int idx = (int)dd;
      float s = c[r] * scale * relk[idx * HH + h] + relb[idx * HH + h];
      out[r] = (ok == 0) ? NEGV : s;
    }
  };

  // ---- pass 1: online row max + exp-sum ----
  float mrow[8], lrow[8];
#pragma unroll
  for (int r = 0; r < 8; ++r) { mrow[r] = -3.0e38f; lrow[r] = 0.0f; }
  for (int j0 = 0; j0 < SS; j0 += 16) {
    float st[8];
    score_tile(j0, st);
#pragma unroll
    for (int r = 0; r < 8; ++r) {
      float t = st[r];
      for (int off = 1; off < 16; off <<= 1) t = fmaxf(t, __shfl_xor(t, off, 32));
      float nm = fmaxf(mrow[r], t);
      float e  = __expf(st[r] - nm);
      for (int off = 1; off < 16; off <<= 1) e += __shfl_xor(e, off, 32);
      lrow[r] = lrow[r] * __expf(mrow[r] - nm) + e;
      mrow[r] = nm;
    }
  }
  float linv[8];
#pragma unroll
  for (int r = 0; r < 8; ++r) linv[r] = 1.0f / lrow[r];

  // ---- pass 2: normalized P -> asum (+=), P@V via WMMA ----
  v8f acc0 = {}, acc1 = {};
  for (int j0 = 0; j0 < SS; j0 += 32) {
    float st0[8], st1[8];
    score_tile(j0, st0);
    score_tile(j0 + 16, st1);
#pragma unroll
    for (int r = 0; r < 8; ++r) {
      float p0 = __expf(st0[r] - mrow[r]) * linv[r];
      float p1 = __expf(st1[r] - mrow[r]) * linv[r];
      size_t base = (bhS + i0 + r + 8 * half) * (size_t)SS + j0;
      if (first) {
        asum[base + nn]      = p0 * 0.25f;
        asum[base + 16 + nn] = p1 * 0.25f;
      } else {
        asum[base + nn]      += p0 * 0.25f;
        asum[base + 16 + nn] += p1 * 0.25f;
      }
      plds[wave][(r + 8 * half) * 32 + nn]      = (__bf16)p0;
      plds[wave][(r + 8 * half) * 32 + 16 + nn] = (__bf16)p1;
    }
    asm volatile("s_wait_dscnt 0x0" ::: "memory");  // CDNA5 split DS counter
    bf16x16 pf;
#pragma unroll
    for (int v = 0; v < 8; ++v) {  // A-layout re-read, 32-bit LDS reads
      int kk = 2 * (v & 3) + 16 * (v >> 2) + 8 * half;
      bf16x2 t = *(const bf16x2*)(&plds[wave][nn * 32 + kk]);
      pf[2 * v]     = t[0];
      pf[2 * v + 1] = t[1];
    }
    const __bf16* vrow = vp + (bhS + j0) * DKK + nn;
    bf16x16 vf0, vf1;
#pragma unroll
    for (int s = 0; s < 16; ++s) {
      int jk = 8 * half + s + (s & 8);
      vf0[s] = vrow[jk * DKK];
      vf1[s] = vrow[jk * DKK + 16];
    }
    acc0 = wmma_bf16(pf, vf0, acc0);
    acc1 = wmma_bf16(pf, vf1, acc1);
  }
#pragma unroll
  for (int r = 0; r < 8; ++r) {
    __bf16* crow = ctxb + ((size_t)(b * SS + i0 + r + 8 * half) * HH + h) * DKK;
    crow[nn]      = (__bf16)acc0[r];
    crow[16 + nn] = (__bf16)acc1[r];
  }
}

// ---------------- LayerNorm over F=256 ----------------
__global__ __launch_bounds__(256) void ln_kernel(
    const float* __restrict__ in, const float* __restrict__ g,
    const float* __restrict__ bb, float* __restrict__ outf,
    __bf16* __restrict__ outb) {
  __shared__ float red[256];
  int row = blockIdx.x, t = threadIdx.x;
  float x = in[(size_t)row * FF + t];
  red[t] = x;
  __syncthreads();
  for (int s = 128; s > 0; s >>= 1) { if (t < s) red[t] += red[t + s]; __syncthreads(); }
  float mu = red[0] * (1.0f / FF);
  __syncthreads();
  float dx = x - mu;
  red[t] = dx * dx;
  __syncthreads();
  for (int s = 128; s > 0; s >>= 1) { if (t < s) red[t] += red[t + s]; __syncthreads(); }
  float var = red[0] * (1.0f / FF);
  float y = dx * rsqrtf(var + 1e-5f) * g[t] + bb[t];
  if (outf) outf[(size_t)row * FF + t] = y;
  if (outb) outb[(size_t)row * FF + t] = (__bf16)y;
}

// ---------------- host driver ----------------
extern "C" void kernel_launch(void* const* d_in, const int* in_sizes, int n_in,
                              void* d_out, int out_size, void* d_ws, size_t ws_size,
                              hipStream_t stream) {
  const float* x_in  = (const float*)d_in[0];
  const int*   maskp = (const int*)d_in[1];
  const int*   pos   = (const int*)d_in[2];
  const float* Wq = (const float*)d_in[3];   const float* bq = (const float*)d_in[4];
  const float* Wk = (const float*)d_in[5];   const float* bk = (const float*)d_in[6];
  const float* Wv = (const float*)d_in[7];   const float* bv = (const float*)d_in[8];
  const float* Wo = (const float*)d_in[9];   const float* bo = (const float*)d_in[10];
  const float* relk = (const float*)d_in[11]; const float* relb = (const float*)d_in[12];
  const float* ln1g = (const float*)d_in[13]; const float* ln1b = (const float*)d_in[14];
  const float* W1 = (const float*)d_in[15];  const float* b1 = (const float*)d_in[16];
  const float* W2 = (const float*)d_in[17];  const float* b2 = (const float*)d_in[18];
  const float* ln2g = (const float*)d_in[19]; const float* ln2b = (const float*)d_in[20];

  float* out_x    = (float*)d_out;
  float* out_attn = out_x + (size_t)BS * FF;

  // workspace carve-out (256B aligned chunks)
  char* cur = (char*)d_ws;
  auto alloc = [&](size_t bytes) {
    char* p = cur;
    cur += (bytes + 255) & ~(size_t)255;
    return (void*)p;
  };
  __bf16* xb   = (__bf16*)alloc((size_t)BS * FF * 2);
  __bf16* qp   = (__bf16*)alloc((size_t)BS * HD * 2);
  __bf16* kt   = (__bf16*)alloc((size_t)BS * HD * 2);
  __bf16* vp   = (__bf16*)alloc((size_t)BS * HD * 2);
  __bf16* ctxb = (__bf16*)alloc((size_t)BS * HD * 2);
  __bf16* h1b  = (__bf16*)alloc((size_t)BS * FF * 2);
  __bf16* hidb = (__bf16*)alloc((size_t)BS * FFN * 2);
  float*  projf = (float*)alloc((size_t)BS * FF * 4);
  float*  ffnf  = (float*)alloc((size_t)BS * FF * 4);
  __bf16* wqb = (__bf16*)alloc((size_t)LAYERS * FF * HD * 2);
  __bf16* wkb = (__bf16*)alloc((size_t)LAYERS * FF * HD * 2);
  __bf16* wvb = (__bf16*)alloc((size_t)LAYERS * FF * HD * 2);
  __bf16* wob = (__bf16*)alloc((size_t)LAYERS * HD * FF * 2);
  __bf16* w1b = (__bf16*)alloc((size_t)LAYERS * FF * FFN * 2);
  __bf16* w2b = (__bf16*)alloc((size_t)LAYERS * FFN * FF * 2);

  auto cvt = [&](const float* src, __bf16* dst, int n) {
    cvt_kernel<<<(n + 255) / 256, 256, 0, stream>>>(src, dst, n);
  };
  cvt(x_in, xb, BS * FF);
  cvt(Wq, wqb, LAYERS * FF * HD);
  cvt(Wk, wkb, LAYERS * FF * HD);
  cvt(Wv, wvb, LAYERS * FF * HD);
  cvt(Wo, wob, LAYERS * HD * FF);
  cvt(W1, w1b, LAYERS * FF * FFN);
  cvt(W2, w2b, LAYERS * FFN * FF);

  dim3 blk(256);
  dim3 g256(HD / 64, BS / 128);    // 16x64 strips
  dim3 g1024(FFN / 64, BS / 128);
  dim3 gattn(SS / 64, BBATCH * HH);

  for (int l = 0; l < LAYERS; ++l) {
    const __bf16* wq = wqb + (size_t)l * FF * HD;  const float* bq_ = bq + l * HD;
    const __bf16* wk = wkb + (size_t)l * FF * HD;  const float* bk_ = bk + l * HD;
    const __bf16* wv = wvb + (size_t)l * FF * HD;  const float* bv_ = bv + l * HD;
    const __bf16* wo = wob + (size_t)l * HD * FF;  const float* bo_ = bo + l * FF;
    const __bf16* w1 = w1b + (size_t)l * FF * FFN; const float* b1_ = b1 + l * FFN;
    const __bf16* w2 = w2b + (size_t)l * FFN * FF; const float* b2_ = b2 + l * FF;

    // QKV projections writing WMMA-fragment-friendly bf16 layouts directly
    gemm_kernel<<<g256, blk, 0, stream>>>(xb, wq, bq_, nullptr, qp, BS, HD, FF, 8);
    gemm_kernel<<<g256, blk, 0, stream>>>(xb, wk, bk_, nullptr, kt, BS, HD, FF, 16);
    gemm_kernel<<<g256, blk, 0, stream>>>(xb, wv, bv_, nullptr, vp, BS, HD, FF, 8);

    attn_kernel<<<gattn, 128, 0, stream>>>(qp, kt, vp, pos, maskp,
                                           relk + l * REL * HH, relb + l * REL * HH,
                                           ctxb, out_attn, (l == 0) ? 1 : 0);

    gemm_kernel<<<g256, blk, 0, stream>>>(ctxb, wo, bo_, projf, nullptr, BS, FF, HD, 2);
    ln_kernel<<<BS, 256, 0, stream>>>(projf, ln1g + l * FF, ln1b + l * FF,
                                      nullptr, h1b);
    gemm_kernel<<<g1024, blk, 0, stream>>>(h1b, w1, b1_, nullptr, hidb, BS, FFN, FF, 5);
    gemm_kernel<<<g256, blk, 0, stream>>>(hidb, w2, b2_, ffnf, nullptr, BS, FF, FFN, 2);
    ln_kernel<<<BS, 256, 0, stream>>>(ffnf, ln2g + l * FF, ln2b + l * FF,
                                      (l == LAYERS - 1) ? out_x : nullptr, xb);
  }
}
